// MPRM_Perspective_1589137900241
// MI455X (gfx1250) — compile-verified, matrix operand order checked by reading, not run on hardware
//
#include <hip/hip_runtime.h>

// ---------------- constants ----------------
#define NMOD 4
#define BB   2
#define EE   128
#define NT   4096
#define DI   512
#define DBL_LD 48   /* padded xproj width (logical 40) */
#define EPS  1e-5f

typedef __attribute__((ext_vector_type(16))) __bf16 v16bf;
typedef __attribute__((ext_vector_type(8)))  __bf16 v8bf;
typedef __attribute__((ext_vector_type(8)))  float  v8f;

static __device__ __forceinline__ __bf16 f2bf(float f) {
  unsigned u = __builtin_bit_cast(unsigned, f);
  unsigned r = u + 0x7FFFu + ((u >> 16) & 1u);
  unsigned short s = (unsigned short)(r >> 16);
  return __builtin_bit_cast(__bf16, s);
}
static __device__ __forceinline__ float siluf(float x) { return x / (1.f + __expf(-x)); }
static __device__ __forceinline__ float softplusf(float x) { return x > 20.f ? x : log1pf(__expf(x)); }

// ---------------- bf16 WMMA GEMM ----------------
// D[M x N] = A[M x K] (bf16, row-major, lda) * Wt[N x K]^T (bf16, row-major, ldw)
// N is always padded to a multiple of 16 (ldc = padded column stride).
// Store modes: f32 C (ldc), bf16 Cbf (ldc), or scatter-accumulate into the
// channel-major output volume (m = b*32768+vox, out[(b*128+n)*32768+vox] += v).
// One wave computes one 16x16 tile; inner loop = 4x global_load_b128 + 1 WMMA.
__global__ void k_gemm(const __bf16* __restrict__ A, int lda,
                       const __bf16* __restrict__ Wt, int ldw,
                       int K, int ldc,
                       float* __restrict__ C, const float* __restrict__ bias,
                       int accumulate, int scatter,
                       float* __restrict__ outvol, __bf16* __restrict__ Cbf) {
  int lane = threadIdx.x;
  int m0 = blockIdx.x * 16;
  int n0 = blockIdx.y * 16;
  const __bf16* arow = A + (size_t)(m0 + (lane & 15)) * lda + ((lane >> 4) * 8);
  const __bf16* brow = Wt + (size_t)(n0 + (lane & 15)) * ldw + ((lane >> 4) * 8);

  v8f c;
  if (accumulate) {
#pragma unroll
    for (int r = 0; r < 8; ++r)
      c[r] = C[(size_t)(m0 + ((lane >> 4) * 8) + r) * ldc + n0 + (lane & 15)];
  } else {
#pragma unroll
    for (int r = 0; r < 8; ++r) c[r] = 0.f;
  }

  for (int k0 = 0; k0 < K; k0 += 32) {
    v8bf a0 = *(const v8bf*)(arow + k0);
    v8bf a1 = *(const v8bf*)(arow + k0 + 16);
    v8bf b0 = *(const v8bf*)(brow + k0);
    v8bf b1 = *(const v8bf*)(brow + k0 + 16);
    v16bf af = __builtin_shufflevector(a0, a1, 0, 1, 2, 3, 4, 5, 6, 7,
                                       8, 9, 10, 11, 12, 13, 14, 15);
    v16bf bfv = __builtin_shufflevector(b0, b1, 0, 1, 2, 3, 4, 5, 6, 7,
                                        8, 9, 10, 11, 12, 13, 14, 15);
    c = __builtin_amdgcn_wmma_f32_16x16x32_bf16(false, af, false, bfv, (short)0, c, false, false);
  }

  int n = n0 + (lane & 15);
#pragma unroll
  for (int r = 0; r < 8; ++r) {
    int m = m0 + ((lane >> 4) * 8) + r;
    float v = c[r] + (bias ? bias[n] : 0.f);
    if (scatter) {
      int b = m >> 15;
      int vox = m & 32767;
      outvol[((size_t)(b * 128 + n)) * 32768 + vox] += v;
    } else if (Cbf) {
      Cbf[(size_t)m * ldc + n] = f2bf(v);
    } else {
      C[(size_t)m * ldc + n] = v;
    }
  }
}

// ---------------- converters ----------------
__global__ void k_cvt(const float* __restrict__ src, __bf16* __restrict__ dst, int n) {
  int id = blockIdx.x * 256 + threadIdx.x;
  if (id < n) dst[id] = f2bf(src[id]);
}
// Wt[n*K + k] = bf16(W[k*N + n]) for n < N, else 0 (pad to Npad rows)
__global__ void k_cvtT(const float* __restrict__ W, __bf16* __restrict__ Wt,
                       int K, int N, int Npad) {
  int id = blockIdx.x * 256 + threadIdx.x;
  if (id >= Npad * K) return;
  int k = id % K;
  int n = id / K;
  Wt[id] = (n < N) ? f2bf(W[(size_t)k * N + n]) : f2bf(0.f);
}

// ---------------- patchify + in_proj ----------------
__global__ void k_patch(const float* __restrict__ x0, const float* __restrict__ x1,
                        const float* __restrict__ x2, const float* __restrict__ x3,
                        const float* __restrict__ inw, const float* __restrict__ inb,
                        float* __restrict__ z) {
  int blk = blockIdx.x;
  int n = blk % NT;
  int b = (blk / NT) % BB;
  int i = blk / (NT * BB);
  const float* x = (i == 0) ? x0 : (i == 1) ? x1 : (i == 2) ? x2 : x3;
  __shared__ float p[8];
  int e = threadIdx.x;
  if (e < 8) {
    int gd = n >> 8, gh = (n >> 4) & 15, gw = n & 15;
    int pd = (e >> 2) & 1, ph = (e >> 1) & 1, pw = e & 1;
    p[e] = x[((size_t)(b * 32 + (gd * 2 + pd)) * 32 + (gh * 2 + ph)) * 32 + (gw * 2 + pw)];
  }
  __syncthreads();
  float acc = inb[i * EE + e];
#pragma unroll
  for (int q = 0; q < 8; ++q) acc += p[q] * inw[(i * 8 + q) * EE + e];
  z[(((size_t)(i * BB + b)) * NT + n) * EE + e] = acc;
}

// ---------------- descriptor mean ----------------
__global__ void k_desc(const float* __restrict__ z, float* __restrict__ dsc) {
  int blk = blockIdx.x;  // i*B+b
  int e = threadIdx.x;
  const float* zb = z + ((size_t)blk * NT) * EE;
  float s = 0.f;
  for (int n = 0; n < NT; ++n) s += zb[(size_t)n * EE + e];
  dsc[blk * EE + e] = s / (float)NT;
}

// ---------------- film params ----------------
__global__ void k_film(const float* __restrict__ dsc,
                       const float* __restrict__ gw, const float* __restrict__ gb,
                       const float* __restrict__ bw, const float* __restrict__ bb2,
                       float* __restrict__ fg, float* __restrict__ fb) {
  int blk = blockIdx.x;  // i*B+b
  int i = blk >> 1;
  int e = threadIdx.x;
  __shared__ float ds[EE];
  ds[e] = dsc[blk * EE + e];
  __syncthreads();
  float ag = gb[i * EE + e], ab = bb2[i * EE + e];
  for (int c2 = 0; c2 < EE; ++c2) {
    float dv = ds[c2];
    ag += dv * gw[((size_t)i * EE + c2) * EE + e];
    ab += dv * bw[((size_t)i * EE + c2) * EE + e];
  }
  fg[blk * EE + e] = ag;
  fb[blk * EE + e] = ab;
}

// ---------------- descriptor attention ----------------
__global__ void k_amat(const float* __restrict__ dsc, float* __restrict__ am) {
  int t = threadIdx.x;  // 32 threads: b*16 + i*4 + j
  int j = t & 3, i = (t >> 2) & 3, b = t >> 4;
  float lg = 0.f;
  for (int e = 0; e < EE; ++e)
    lg += dsc[(i * BB + b) * EE + e] * dsc[(j * BB + b) * EE + e];
  __shared__ float L[32];
  L[t] = lg;
  __syncthreads();
  int base = t & ~3;
  float mx = fmaxf(fmaxf(L[base], L[base + 1]), fmaxf(L[base + 2], L[base + 3]));
  float den = __expf(L[base] - mx) + __expf(L[base + 1] - mx) +
              __expf(L[base + 2] - mx) + __expf(L[base + 3] - mx);
  am[b * 16 + i * 4 + j] = __expf(lg - mx) / den;
}

// ---------------- low-rank U,V ----------------
__global__ void k_uv(const float* __restrict__ dsc,
                     const float* __restrict__ ruw, const float* __restrict__ rub,
                     const float* __restrict__ rvw, const float* __restrict__ rvb,
                     float* __restrict__ U, float* __restrict__ V) {
  int id = blockIdx.x * 256 + threadIdx.x;  // 32768
  int r = id & 31;
  int e = (id >> 5) & 127;
  int b = (id >> 12) & 1;
  int j = id >> 13;
  __shared__ float ds[EE];
  if (threadIdx.x < EE) ds[threadIdx.x] = dsc[(j * BB + b) * EE + threadIdx.x];
  __syncthreads();
  int col = e * 32 + r;
  float au = rub[j * 4096 + col], av = rvb[j * 4096 + col];
  for (int c2 = 0; c2 < EE; ++c2) {
    float dv = ds[c2];
    au += dv * ruw[((size_t)j * EE + c2) * 4096 + col];
    av += dv * rvw[((size_t)j * EE + c2) * 4096 + col];
  }
  U[id] = au;
  V[id] = av;
}

// ---------------- Cj = U V^T ----------------
__global__ void k_cj(const float* __restrict__ U, const float* __restrict__ V,
                     float* __restrict__ Cj) {
  int id = blockIdx.x * 256 + threadIdx.x;  // 131072
  int f = id & 127;
  int e = (id >> 7) & 127;
  int b = (id >> 14) & 1;
  int j = id >> 15;
  const float* Ur = U + (((j * BB + b) * EE) + e) * 32;
  const float* Vr = V + (((j * BB + b) * EE) + f) * 32;
  float s = 0.f;
#pragma unroll
  for (int r = 0; r < 32; ++r) s += Ur[r] * Vr[r];
  Cj[id] = s;
}

// ---------------- M_i = sum_j A[b,i,j] C_j ----------------
__global__ void k_mmat(const float* __restrict__ am, const float* __restrict__ Cj,
                       float* __restrict__ Mm) {
  int id = blockIdx.x * 256 + threadIdx.x;  // 131072
  int f = id & 127;
  int e = (id >> 7) & 127;
  int b = (id >> 14) & 1;
  int i = id >> 15;
  float s = 0.f;
#pragma unroll
  for (int j = 0; j < 4; ++j)
    s += am[b * 16 + i * 4 + j] * Cj[((((size_t)(j * BB + b) * EE) + e) * EE) + f];
  Mm[id] = s;
}

// ---------------- LN + FiLM -> u (bf16) ----------------
__global__ void k_ln_film_u(int i, const float* __restrict__ z,
                            const float* __restrict__ fg, const float* __restrict__ fb,
                            const float* __restrict__ lng, const float* __restrict__ lnb,
                            __bf16* __restrict__ uo) {
  int blk = blockIdx.x;  // b*NT+n
  int b = blk / NT;
  int n = blk % NT;
  int e = threadIdx.x;
  float v = z[(((size_t)(i * BB + b)) * NT + n) * EE + e];
  __shared__ float r1[EE], r2[EE];
  r1[e] = v;
  r2[e] = v * v;
  __syncthreads();
  for (int o = 64; o > 0; o >>= 1) {
    if (e < o) { r1[e] += r1[e + o]; r2[e] += r2[e + o]; }
    __syncthreads();
  }
  float mu = r1[0] / (float)EE;
  float var = r2[0] / (float)EE - mu * mu;
  float rs = rsqrtf(var + EPS);
  float s = (v - mu) * rs * lng[i * EE + e] + lnb[i * EE + e];
  float u = fg[(i * BB + b) * EE + e] * s + fb[(i * BB + b) * EE + e];
  uo[((size_t)(b * NT + n)) * EE + e] = f2bf(u);
}

// ---------------- depthwise causal conv + silu ----------------
__global__ void k_conv(int i, const float* __restrict__ xz,
                       const float* __restrict__ cw, const float* __restrict__ cb,
                       float* __restrict__ xc, __bf16* __restrict__ xcbf) {
  int id = blockIdx.x * 256 + threadIdx.x;  // 4,194,304 = (b,n,d)
  int d = id & 511;
  int n = (id >> 9) & 4095;
  int b = id >> 21;
  float acc = cb[i * DI + d];
#pragma unroll
  for (int t = 0; t < 4; ++t) {
    int nn = n - 3 + t;
    if (nn >= 0)
      acc += xz[(((size_t)(b * NT + nn)) << 10) + d] * cw[(i * DI + d) * 4 + t];
  }
  float v = siluf(acc);
  xc[id] = v;
  xcbf[id] = f2bf(v);
}

// ---------------- dt = softplus(dbl[:,:8] @ dtw + dtb) ----------------
__global__ void k_dt(int i, const float* __restrict__ dbl,
                     const float* __restrict__ dtw, const float* __restrict__ dtb2,
                     float* __restrict__ dt) {
  int id = blockIdx.x * 256 + threadIdx.x;
  int d = id & 511;
  int n = (id >> 9) & 4095;
  int b = id >> 21;
  const float* dr = dbl + ((size_t)(b * NT + n)) * DBL_LD;
  float acc = dtb2[i * DI + d];
#pragma unroll
  for (int r = 0; r < 8; ++r) acc += dr[r] * dtw[(i * 8 + r) * DI + d];
  dt[id] = softplusf(acc);
}

// ---------------- scan pass 1: per-chunk (prodA, h) ----------------
__global__ void k_scan1(int i, const float* __restrict__ dt, const float* __restrict__ xc,
                        const float* __restrict__ dbl, const float* __restrict__ Alog,
                        float* __restrict__ aa, float* __restrict__ hh) {
  int id = blockIdx.x * 256 + threadIdx.x;  // 2,097,152 = (b,d,s,ch)
  int ch = id & 127;
  int s = (id >> 7) & 15;
  int d = (id >> 11) & 511;
  int b = id >> 20;
  float negA = -__expf(Alog[(size_t)i * 8192 + d * 16 + s]);
  float a = 1.f, h = 0.f;
  int l0 = ch * 32;
  for (int l = l0; l < l0 + 32; ++l) {
    size_t t = (size_t)(b * NT + l);
    float dtv = dt[t * 512 + d];
    float xcv = xc[t * 512 + d];
    float Bv = dbl[t * DBL_LD + 8 + s];
    float dA = __expf(negA * dtv);
    h = dA * h + dtv * xcv * Bv;
    a *= dA;
  }
  aa[id] = a;
  hh[id] = h;
}

// ---------------- scan pass 2: prefix over chunks ----------------
__global__ void k_scan2(const float* __restrict__ aa, float* __restrict__ hh) {
  int id = blockIdx.x * 256 + threadIdx.x;  // 16384 = (b,d,s)
  float run = 0.f;
  size_t base = (size_t)id * 128;
  for (int ch = 0; ch < 128; ++ch) {
    float a = aa[base + ch];
    float h = hh[base + ch];
    hh[base + ch] = run;   // incoming state for this chunk
    run = a * run + h;
  }
}

// ---------------- scan pass 3: re-sweep, emit y ----------------
__global__ void k_scan3(int i, const float* __restrict__ dt, const float* __restrict__ xc,
                        const float* __restrict__ dbl, const float* __restrict__ Alog,
                        const float* __restrict__ Dp, const float* __restrict__ hh,
                        float* __restrict__ ym) {
  int id = blockIdx.x * 256 + threadIdx.x;  // 131072 = (b,d,ch)
  int ch = id & 127;
  int d = (id >> 7) & 511;
  int b = id >> 16;
  float h[16], negA[16];
#pragma unroll
  for (int s = 0; s < 16; ++s) {
    h[s] = hh[(((size_t)(b * 512 + d) * 16) + s) * 128 + ch];
    negA[s] = -__expf(Alog[(size_t)i * 8192 + d * 16 + s]);
  }
  float Dv = Dp[i * DI + d];
  int l0 = ch * 32;
  for (int l = l0; l < l0 + 32; ++l) {
    size_t t = (size_t)(b * NT + l);
    float dtv = dt[t * 512 + d];
    float xcv = xc[t * 512 + d];
    float dx = dtv * xcv;
    const float* dr = dbl + t * DBL_LD;
    float yv = 0.f;
#pragma unroll
    for (int s = 0; s < 16; ++s) {
      float dA = __expf(negA[s] * dtv);
      h[s] = dA * h[s] + dx * dr[8 + s];
      yv += h[s] * dr[24 + s];
    }
    ym[t * 512 + d] = yv + Dv * xcv;
  }
}

// ---------------- gate: ymbf = bf16(ym * silu(zg)) ----------------
__global__ void k_gate(const float* __restrict__ xz, const float* __restrict__ ym,
                       __bf16* __restrict__ ymbf) {
  int id = blockIdx.x * 256 + threadIdx.x;  // 4,194,304
  int d = id & 511;
  size_t t = (size_t)(id >> 9);
  float zg = xz[(t << 10) + 512 + d];
  ymbf[id] = f2bf(ym[id] * siluf(zg));
}

// ---------------- per-(b,c) token stats for outnorm ----------------
__global__ void k_ystats(const float* __restrict__ y, float* __restrict__ ymu,
                         float* __restrict__ yrs) {
  int b = blockIdx.x;
  int e = threadIdx.x;
  float s = 0.f, s2 = 0.f;
  for (int n = 0; n < NT; ++n) {
    float v = y[((size_t)(b * NT + n)) * EE + e];
    s += v;
    s2 += v * v;
  }
  float mu = s / (float)NT;
  float var = s2 / (float)NT - mu * mu;
  ymu[b * EE + e] = mu;
  yrs[b * EE + e] = rsqrtf(var + EPS);
}

// ---------------- per-(i,b) volume stats ----------------
__global__ void k_xstats(const float* __restrict__ x0, const float* __restrict__ x1,
                         const float* __restrict__ x2, const float* __restrict__ x3,
                         float* __restrict__ xmu, float* __restrict__ xrs) {
  int blk = blockIdx.x;  // i*2+b
  int b = blk & 1;
  int i = blk >> 1;
  const float* x = (i == 0) ? x0 : (i == 1) ? x1 : (i == 2) ? x2 : x3;
  const float* xb = x + (size_t)b * 32768;
  int t = threadIdx.x;
  float s = 0.f, s2 = 0.f;
  for (int k = t; k < 32768; k += 256) {
    float v = xb[k];
    s += v;
    s2 += v * v;
  }
  __shared__ float r1[256], r2[256];
  r1[t] = s;
  r2[t] = s2;
  __syncthreads();
  for (int o = 128; o > 0; o >>= 1) {
    if (t < o) { r1[t] += r1[t + o]; r2[t] += r2[t + o]; }
    __syncthreads();
  }
  if (t == 0) {
    float mu = r1[0] / 32768.f;
    float var = r2[0] / 32768.f - mu * mu;
    xmu[blk] = mu;
    xrs[blk] = rsqrtf(var + EPS);
  }
}

// ---------------- val = bf16(silu(bys) * outnorm(upsample(y))) ----------------
__global__ void k_val(int i, const float* __restrict__ x, const float* __restrict__ y,
                      const float* __restrict__ ymu, const float* __restrict__ yrs,
                      const float* __restrict__ xmu, const float* __restrict__ xrs,
                      const float* __restrict__ byng, const float* __restrict__ bynb,
                      const float* __restrict__ byc1w, const float* __restrict__ byc1b,
                      const float* __restrict__ byc2w, const float* __restrict__ byc2b,
                      const float* __restrict__ og, const float* __restrict__ ob,
                      __bf16* __restrict__ valbf) {
  int id = blockIdx.x * 256 + threadIdx.x;  // 8,388,608 = (b,vox,c)
  int c = id & 127;
  int vox = (id >> 7) & 32767;
  int b = id >> 22;
  int dz = vox >> 10, hy = (vox >> 5) & 31, wx = vox & 31;
  int n = ((dz >> 1) * 16 + (hy >> 1)) * 16 + (wx >> 1);
  float yv = y[((size_t)(b * NT + n)) * EE + c];
  float nv = (yv - ymu[b * EE + c]) * yrs[b * EE + c] * og[c] + ob[c];
  float xv = x[(size_t)b * 32768 + vox];
  float uv = (xv - xmu[i * 2 + b]) * xrs[i * 2 + b] * byng[i] + bynb[i];
  uv = uv * byc1w[i] + byc1b[i];
  float bys = byc2w[i * EE + c] * uv + byc2b[i * EE + c];
  valbf[id] = f2bf(siluf(bys) * nv);
}

__global__ void k_zero(float* __restrict__ p, int nelem) {
  int id = blockIdx.x * 256 + threadIdx.x;
  if (id < nelem) p[id] = 0.f;
}

// ---------------- host launch ----------------
extern "C" void kernel_launch(void* const* d_in, const int* in_sizes, int n_in,
                              void* d_out, int out_size, void* d_ws, size_t ws_size,
                              hipStream_t stream) {
  (void)in_sizes; (void)n_in; (void)out_size; (void)ws_size;
  const float* xs[4] = {(const float*)d_in[0], (const float*)d_in[1],
                        (const float*)d_in[2], (const float*)d_in[3]};
  const float* in_proj_w = (const float*)d_in[4];
  const float* in_proj_b = (const float*)d_in[5];
  const float* ln_g      = (const float*)d_in[6];
  const float* ln_b      = (const float*)d_in[7];
  const float* film_gw   = (const float*)d_in[8];
  const float* film_gb   = (const float*)d_in[9];
  const float* film_bw   = (const float*)d_in[10];
  const float* film_bb   = (const float*)d_in[11];
  const float* m_inw     = (const float*)d_in[12];
  const float* m_convw   = (const float*)d_in[13];
  const float* m_convb   = (const float*)d_in[14];
  const float* m_xprojw  = (const float*)d_in[15];
  const float* m_dtw     = (const float*)d_in[16];
  const float* m_dtb     = (const float*)d_in[17];
  const float* m_Alog    = (const float*)d_in[18];
  const float* m_D       = (const float*)d_in[19];
  const float* m_outw    = (const float*)d_in[20];
  const float* ru_w      = (const float*)d_in[21];
  const float* ru_b      = (const float*)d_in[22];
  const float* rv_w      = (const float*)d_in[23];
  const float* rv_b      = (const float*)d_in[24];
  const float* dlin_w    = (const float*)d_in[25];
  const float* dlin_b    = (const float*)d_in[26];
  const float* outnorm_g = (const float*)d_in[27];
  const float* outnorm_b = (const float*)d_in[28];
  const float* byn_g     = (const float*)d_in[29];
  const float* byn_b     = (const float*)d_in[30];
  const float* byc1_w    = (const float*)d_in[31];
  const float* byc1_b    = (const float*)d_in[32];
  const float* byc2_w    = (const float*)d_in[33];
  const float* byc2_b    = (const float*)d_in[34];
  const float* post_w    = (const float*)d_in[35];
  const float* post_b    = (const float*)d_in[36];

  float* out = (float*)d_out;
  float* ws = (float*)d_ws;
  size_t off = 0;
  auto alloc_f = [&](size_t n) -> float* {
    float* p = ws + off;
    off += (n + 3) & ~(size_t)3;
    return p;
  };
  auto alloc_h = [&](size_t n) -> __bf16* {
    __bf16* p = (__bf16*)(ws + off);
    off += (((n + 1) / 2) + 3) & ~(size_t)3;
    return p;
  };

  // f32 buffers
  float* z    = alloc_f((size_t)NMOD * BB * NT * EE);   // tokens, persistent
  float* dsc  = alloc_f(NMOD * BB * EE);
  float* fg   = alloc_f(NMOD * BB * EE);
  float* fb   = alloc_f(NMOD * BB * EE);
  float* am   = alloc_f(32);
  float* Ub   = alloc_f(NMOD * BB * EE * 32);
  float* Vb   = alloc_f(NMOD * BB * EE * 32);
  float* Cjb  = alloc_f(NMOD * BB * EE * EE);
  float* Mm   = alloc_f(NMOD * BB * EE * EE);
  float* xz   = alloc_f((size_t)BB * NT * 1024);        // reused as val_bf in phase 3
  float* xcb  = alloc_f((size_t)BB * NT * DI);
  float* dblb = alloc_f((size_t)BB * NT * DBL_LD);
  float* dtb  = alloc_f((size_t)BB * NT * DI);
  float* aab  = alloc_f((size_t)BB * DI * 16 * 128);
  float* hhb  = alloc_f((size_t)BB * DI * 16 * 128);
  float* ymb  = alloc_f((size_t)BB * NT * DI);
  float* ybuf = alloc_f((size_t)BB * NT * EE);
  float* ymu  = alloc_f(BB * EE);
  float* yrs  = alloc_f(BB * EE);
  float* xmu  = alloc_f(8);
  float* xrs  = alloc_f(8);

  // bf16 buffers (GEMM operands, fragment-friendly layouts)
  __bf16* z_bf  = alloc_h((size_t)NMOD * BB * NT * EE);
  __bf16* Hbf   = alloc_h((size_t)NMOD * BB * NT * EE);
  __bf16* ub_bf = alloc_h((size_t)BB * NT * EE);
  __bf16* xcbf  = alloc_h((size_t)BB * NT * DI);
  __bf16* ymbf  = alloc_h((size_t)BB * NT * DI);
  __bf16* wt_in   = alloc_h(1024 * 128);
  __bf16* wt_xp   = alloc_h(DBL_LD * 512);
  __bf16* wt_out  = alloc_h(128 * 512);
  __bf16* wt_dlin = alloc_h(128 * 128);
  __bf16* wt_post = alloc_h(128 * 128);
  __bf16* wt_mm   = alloc_h(NMOD * BB * 128 * 128);
  __bf16* val_bf  = (__bf16*)xz;  // xz dead in phase 3

  // ---- phase 1: tokens, descriptors, small matrices ----
  k_patch<<<NMOD * BB * NT, 128, 0, stream>>>(xs[0], xs[1], xs[2], xs[3], in_proj_w, in_proj_b, z);
  k_cvt<<<16384, 256, 0, stream>>>(z, z_bf, NMOD * BB * NT * EE);
  k_desc<<<NMOD * BB, 128, 0, stream>>>(z, dsc);
  k_film<<<NMOD * BB, 128, 0, stream>>>(dsc, film_gw, film_gb, film_bw, film_bb, fg, fb);
  k_amat<<<1, 32, 0, stream>>>(dsc, am);
  k_uv<<<128, 256, 0, stream>>>(dsc, ru_w, ru_b, rv_w, rv_b, Ub, Vb);
  k_cj<<<512, 256, 0, stream>>>(Ub, Vb, Cjb);
  k_mmat<<<512, 256, 0, stream>>>(am, Cjb, Mm);
  for (int s = 0; s < NMOD * BB; ++s)
    k_cvtT<<<64, 256, 0, stream>>>(Mm + (size_t)s * EE * EE, wt_mm + (size_t)s * EE * EE,
                                   128, 128, 128);
  k_xstats<<<8, 256, 0, stream>>>(xs[0], xs[1], xs[2], xs[3], xmu, xrs);
  k_zero<<<32768, 256, 0, stream>>>(out, BB * EE * 32768);

  // ---- phase 2: per-module mamba chain ----
  for (int i = 0; i < NMOD; ++i) {
    // weight transposes (f32 -> bf16, N x K layout)
    k_cvtT<<<512, 256, 0, stream>>>(m_inw + (size_t)i * EE * 1024, wt_in, 128, 1024, 1024);
    k_cvtT<<<96, 256, 0, stream>>>(m_xprojw + (size_t)i * DI * 40, wt_xp, 512, 40, DBL_LD);
    k_cvtT<<<256, 256, 0, stream>>>(m_outw + (size_t)i * DI * EE, wt_out, 512, 128, 128);

    k_ln_film_u<<<BB * NT, 128, 0, stream>>>(i, z, fg, fb, ln_g, ln_b, ub_bf);
    // xz = u @ m_inw  (8192 x 1024, K=128)
    k_gemm<<<dim3(512, 64), 32, 0, stream>>>(ub_bf, EE, wt_in, EE, EE, 1024,
                                             xz, nullptr, 0, 0, nullptr, nullptr);
    k_conv<<<16384, 256, 0, stream>>>(i, xz, m_convw, m_convb, xcb, xcbf);
    // dbl = xc @ xproj (8192 x 48pad, K=512)
    k_gemm<<<dim3(512, 3), 32, 0, stream>>>(xcbf, DI, wt_xp, DI, DI, DBL_LD,
                                            dblb, nullptr, 0, 0, nullptr, nullptr);
    k_dt<<<16384, 256, 0, stream>>>(i, dblb, m_dtw, m_dtb, dtb);
    k_scan1<<<8192, 256, 0, stream>>>(i, dtb, xcb, dblb, m_Alog, aab, hhb);
    k_scan2<<<64, 256, 0, stream>>>(aab, hhb);
    k_scan3<<<512, 256, 0, stream>>>(i, dtb, xcb, dblb, m_Alog, m_D, hhb, ymb);
    k_gate<<<16384, 256, 0, stream>>>(xz, ymb, ymbf);
    // H_i = ym @ outw (8192 x 128, K=512) -> bf16 directly
    k_gemm<<<dim3(512, 8), 32, 0, stream>>>(ymbf, DI, wt_out, DI, DI, EE,
                                            nullptr, nullptr, 0, 0, nullptr,
                                            Hbf + (size_t)i * BB * NT * EE);
  }

  // ---- phase 3: fusion into output volume ----
  for (int i = 0; i < NMOD; ++i) {
    k_cvtT<<<64, 256, 0, stream>>>(dlin_w + (size_t)i * EE * EE, wt_dlin, 128, 128, 128);
    k_cvtT<<<64, 256, 0, stream>>>(post_w + (size_t)i * EE * EE, wt_post, 128, 128, 128);

    // y = z_i @ dlin + b
    k_gemm<<<dim3(512, 8), 32, 0, stream>>>(z_bf + (size_t)i * BB * NT * EE, EE,
                                            wt_dlin, EE, EE, EE,
                                            ybuf, dlin_b + i * EE, 0, 0, nullptr, nullptr);
    // y += H_i @ M_i   (M_i per batch)
    for (int b = 0; b < BB; ++b) {
      k_gemm<<<dim3(256, 8), 32, 0, stream>>>(Hbf + ((size_t)i * BB + b) * NT * EE, EE,
                                              wt_mm + ((size_t)(i * BB + b)) * EE * EE, EE,
                                              EE, EE,
                                              ybuf + (size_t)b * NT * EE, nullptr, 1, 0,
                                              nullptr, nullptr);
    }
    k_ystats<<<BB, 128, 0, stream>>>(ybuf, ymu, yrs);
    k_val<<<32768, 256, 0, stream>>>(i, xs[i], ybuf, ymu, yrs, xmu, xrs,
                                     byn_g, byn_b, byc1_w, byc1_b, byc2_w, byc2_b,
                                     outnorm_g, outnorm_b, val_bf);
    // out += val @ post_w + post_b  (scatter to channel-major volume)
    k_gemm<<<dim3(4096, 8), 32, 0, stream>>>(val_bf, EE, wt_post, EE, EE, EE,
                                             nullptr, post_b + i * EE, 0, 1, out, nullptr);
  }
}
